// GNN_graphSage_43095701848157
// MI455X (gfx1250) — compile-verified, matrix operand order checked by reading, not run on hardware
//
#include <hip/hip_runtime.h>
#include <stdint.h>

// ---------------------------------------------------------------------------
// Types for CDNA5 WMMA (wave32, v_wmma_f32_16x16x32_bf16)
// ---------------------------------------------------------------------------
typedef __attribute__((ext_vector_type(16))) __bf16 v16bf;
typedef __attribute__((ext_vector_type(8)))  float  v8f;

union FragBF {
    uint4 u[2];   // 32 bytes = 16 bf16
    v16bf v;
};

__device__ __forceinline__ uint16_t f32_to_bf16(float f) {
    uint32_t u = __builtin_bit_cast(uint32_t, f);
    uint32_t r = u + 0x7FFFu + ((u >> 16) & 1u);   // round-to-nearest-even
    return (uint16_t)(r >> 16);
}
__device__ __forceinline__ float bf16_to_f32(uint16_t h) {
    uint32_t u = ((uint32_t)h) << 16;
    return __builtin_bit_cast(float, u);
}

// Load a 16-element bf16 fragment: two contiguous 16B chunks (K+0..7, K+16..23)
__device__ __forceinline__ v16bf load_frag(const uint16_t* __restrict__ p) {
    FragBF f;
    f.u[0] = *reinterpret_cast<const uint4*>(p);
    f.u[1] = *reinterpret_cast<const uint4*>(p + 16);
    return f.v;
}

// ---------------------------------------------------------------------------
// Edge-count kernel: cnt[dst[e]] += 1
// ---------------------------------------------------------------------------
__global__ void count_kernel(const long long* __restrict__ dst, float* __restrict__ cnt, int E) {
    int e = blockIdx.x * blockDim.x + threadIdx.x;
    if (e < E) atomicAdd(&cnt[(int)dst[e]], 1.0f);
}

// Scatter-add f32 features over edges: agg[dst] += x[src]
__global__ void agg_edges_f32(const float* __restrict__ x,
                              const long long* __restrict__ src,
                              const long long* __restrict__ dst,
                              float* __restrict__ agg, int E, int F) {
    for (int e = blockIdx.x; e < E; e += gridDim.x) {
        size_t sb = (size_t)(int)src[e] * F;
        size_t db = (size_t)(int)dst[e] * F;
        for (int f = threadIdx.x; f < F; f += blockDim.x)
            atomicAdd(&agg[db + f], x[sb + f]);
    }
}

// Scatter-add bf16 features (read bf16, accumulate f32)
__global__ void agg_edges_bf16(const uint16_t* __restrict__ x,
                               const long long* __restrict__ src,
                               const long long* __restrict__ dst,
                               float* __restrict__ agg, int E, int F) {
    for (int e = blockIdx.x; e < E; e += gridDim.x) {
        size_t sb = (size_t)(int)src[e] * F;
        size_t db = (size_t)(int)dst[e] * F;
        for (int f = threadIdx.x; f < F; f += blockDim.x)
            atomicAdd(&agg[db + f], bf16_to_f32(x[sb + f]));
    }
}

// mean = agg / max(cnt,1), converted to bf16; rows >= Nvalid zero-filled (padding)
__global__ void mean_bf16_kernel(const float* __restrict__ agg,
                                 const float* __restrict__ cnt,
                                 uint16_t* __restrict__ out,
                                 int Nvalid, int Mp, int F) {
    size_t total = (size_t)Mp * F;
    for (size_t i = (size_t)blockIdx.x * blockDim.x + threadIdx.x; i < total;
         i += (size_t)gridDim.x * blockDim.x) {
        int row = (int)(i / F);
        float v = 0.f;
        if (row < Nvalid) v = agg[i] / fmaxf(cnt[row], 1.0f);
        out[i] = f32_to_bf16(v);
    }
}

// f32 -> bf16 with padding-row zero-fill
__global__ void cvt_bf16_kernel(const float* __restrict__ x, uint16_t* __restrict__ out,
                                int Nvalid, int Mp, int F) {
    size_t total = (size_t)Mp * F;
    for (size_t i = (size_t)blockIdx.x * blockDim.x + threadIdx.x; i < total;
         i += (size_t)gridDim.x * blockDim.x) {
        int row = (int)(i / F);
        out[i] = f32_to_bf16(row < Nvalid ? x[i] : 0.f);
    }
}

// W [K, Nc] f32 row-major  ->  Wt [Nc, K] bf16 (so B fragments load contiguously)
__global__ void transpose_w_kernel(const float* __restrict__ W, uint16_t* __restrict__ Wt,
                                   int K, int Nc) {
    size_t total = (size_t)K * Nc;
    for (size_t i = (size_t)blockIdx.x * blockDim.x + threadIdx.x; i < total;
         i += (size_t)gridDim.x * blockDim.x) {
        int k = (int)(i / Nc), n = (int)(i % Nc);
        Wt[(size_t)n * K + k] = f32_to_bf16(W[i]);
    }
}

// ---------------------------------------------------------------------------
// Fused SAGE GEMM: out = relu( Am @ WlT^T + As @ WrT^T + bias )
//   Am, As : [Mp, K]  bf16 row-major (Mp multiple of 32)
//   WlT/WrT: [Nc, K]  bf16 (pre-transposed weights)
// Each wave computes a 32(M) x 64(N) tile with v_wmma_f32_16x16x32_bf16.
// Block = 8 waves covering 512 columns; grid = (Mp/32, Nc/512).
// ---------------------------------------------------------------------------
template <bool OUT_BF16>
__global__ __launch_bounds__(256)
void sage_gemm(const uint16_t* __restrict__ Am,
               const uint16_t* __restrict__ As,
               const uint16_t* __restrict__ WlT,
               const uint16_t* __restrict__ WrT,
               const float*    __restrict__ bias,
               void*           __restrict__ outp,
               int K, int Nc, int Mvalid) {
    const int lane   = threadIdx.x & 31;
    const int wave   = threadIdx.x >> 5;
    const int mBase  = blockIdx.x * 32;
    const int nStrip = (blockIdx.y * 8 + wave) * 64;

    const int lrow = lane & 15;              // row within 16 (A) / col within 16 (B)
    const int kb   = (lane < 16) ? 0 : 8;    // K-half select per fragment layout

    const v8f vzero = {0.f, 0.f, 0.f, 0.f, 0.f, 0.f, 0.f, 0.f};
    v8f acc[2][4];
#pragma unroll
    for (int i = 0; i < 2; ++i)
#pragma unroll
        for (int j = 0; j < 4; ++j) acc[i][j] = vzero;

    // --- pass 1: mean features x W_l ; pass 2: self features x W_r -------
#pragma unroll 1
    for (int pass = 0; pass < 2; ++pass) {
        const uint16_t* A  = pass ? As : Am;
        const uint16_t* Wt = pass ? WrT : WlT;
        const uint16_t* a0 = A  + (size_t)(mBase + lrow) * K + kb;
        const uint16_t* a1 = a0 + (size_t)16 * K;
        const uint16_t* bp = Wt + (size_t)(nStrip + lrow) * K + kb;

        for (int k0 = 0; k0 < K; k0 += 32) {
            v16bf fa0 = load_frag(a0 + k0);
            v16bf fa1 = load_frag(a1 + k0);
#pragma unroll
            for (int t = 0; t < 4; ++t) {
                v16bf fb = load_frag(bp + (size_t)(t * 16) * K + k0);
                acc[0][t] = __builtin_amdgcn_wmma_f32_16x16x32_bf16(
                    false, fa0, false, fb, (short)0, acc[0][t], false, false);
                acc[1][t] = __builtin_amdgcn_wmma_f32_16x16x32_bf16(
                    false, fa1, false, fb, (short)0, acc[1][t], false, false);
            }
        }
    }

    // --- epilogue: bias + ReLU + store ------------------------------------
    // C/D layout: VGPR r -> row = r + (lane<16 ? 0 : 8), col = lane & 15
    const int rOff = (lane < 16) ? 0 : 8;
#pragma unroll
    for (int t = 0; t < 4; ++t) {
        int col  = nStrip + t * 16 + lrow;
        float bc = bias[col];
#pragma unroll
        for (int mt = 0; mt < 2; ++mt) {
#pragma unroll
            for (int r = 0; r < 8; ++r) {
                int row = mBase + mt * 16 + rOff + r;
                float v = acc[mt][t][r] + bc;
                v = v > 0.f ? v : 0.f;
                if (OUT_BF16) {
                    ((uint16_t*)outp)[(size_t)row * Nc + col] = f32_to_bf16(v);
                } else {
                    if (row < Mvalid)
                        ((float*)outp)[(size_t)row * Nc + col] = v;
                }
            }
        }
    }
}

// ---------------------------------------------------------------------------
// Row-wise L2 normalize (in place on d_out)
// ---------------------------------------------------------------------------
__global__ __launch_bounds__(256)
void l2norm_kernel(float* __restrict__ out, int F) {
    __shared__ float red[256];
    size_t base = (size_t)blockIdx.x * F;
    float s = 0.f;
    for (int f = threadIdx.x; f < F; f += 256) {
        float v = out[base + f];
        s += v * v;
    }
    red[threadIdx.x] = s;
    __syncthreads();
    for (int st = 128; st > 0; st >>= 1) {
        if (threadIdx.x < st) red[threadIdx.x] += red[threadIdx.x + st];
        __syncthreads();
    }
    float inv = 1.0f / fmaxf(sqrtf(red[0]), 1e-12f);
    for (int f = threadIdx.x; f < F; f += 256) out[base + f] *= inv;
}

// ---------------------------------------------------------------------------
// Host side
// ---------------------------------------------------------------------------
extern "C" void kernel_launch(void* const* d_in, const int* in_sizes, int n_in,
                              void* d_out, int out_size, void* d_ws, size_t ws_size,
                              hipStream_t stream) {
    const float*     x   = (const float*)d_in[0];
    const long long* ei  = (const long long*)d_in[2];
    const float*     W1l = (const float*)d_in[3];
    const float*     W1r = (const float*)d_in[4];
    const float*     b1  = (const float*)d_in[5];
    const float*     W2l = (const float*)d_in[6];
    const float*     W2r = (const float*)d_in[7];
    const float*     b2  = (const float*)d_in[8];
    const float*     W3l = (const float*)d_in[9];
    const float*     W3r = (const float*)d_in[10];
    const float*     b3  = (const float*)d_in[11];

    const int E   = in_sizes[2] / 2;
    const int Fin = 128, H = 2048, Fout = 1024;
    const int N   = in_sizes[0] / Fin;          // 50000
    const int Mp  = (N + 31) & ~31;             // padded rows for 32-row tiles

    const long long* src = ei;
    const long long* dst = ei + E;

    // ---- carve workspace (256B aligned) ----
    char*  ws  = (char*)d_ws;
    size_t off = 0;
    auto carve = [&](size_t bytes) -> void* {
        void* p = ws + off;
        off = (off + bytes + 255) & ~(size_t)255;
        return p;
    };
    float*    cnt    = (float*)   carve((size_t)Mp * 4);
    float*    agg    = (float*)   carve((size_t)Mp * H * 4);
    uint16_t* meanBf = (uint16_t*)carve((size_t)Mp * H * 2);
    uint16_t* hA     = (uint16_t*)carve((size_t)Mp * H * 2);
    uint16_t* hB     = (uint16_t*)carve((size_t)Mp * H * 2);
    uint16_t* xBf    = (uint16_t*)carve((size_t)Mp * Fin * 2);
    uint16_t* W1lT   = (uint16_t*)carve((size_t)H * Fin * 2);
    uint16_t* W1rT   = (uint16_t*)carve((size_t)H * Fin * 2);
    uint16_t* W2lT   = (uint16_t*)carve((size_t)H * H * 2);
    uint16_t* W2rT   = (uint16_t*)carve((size_t)H * H * 2);
    uint16_t* W3lT   = (uint16_t*)carve((size_t)Fout * H * 2);
    uint16_t* W3rT   = (uint16_t*)carve((size_t)Fout * H * 2);

    const int EW = 4096;   // grid for elementwise grid-stride kernels

    // ---- one-time per-launch prep ----
    hipMemsetAsync(cnt, 0, (size_t)Mp * 4, stream);
    count_kernel<<<(E + 255) / 256, 256, 0, stream>>>(dst, cnt, E);
    cvt_bf16_kernel<<<EW, 256, 0, stream>>>(x, xBf, N, Mp, Fin);
    transpose_w_kernel<<<EW, 256, 0, stream>>>(W1l, W1lT, Fin, H);
    transpose_w_kernel<<<EW, 256, 0, stream>>>(W1r, W1rT, Fin, H);
    transpose_w_kernel<<<EW, 256, 0, stream>>>(W2l, W2lT, H, H);
    transpose_w_kernel<<<EW, 256, 0, stream>>>(W2r, W2rT, H, H);
    transpose_w_kernel<<<EW, 256, 0, stream>>>(W3l, W3lT, H, Fout);
    transpose_w_kernel<<<EW, 256, 0, stream>>>(W3r, W3rT, H, Fout);

    // ---- layer 1: 128 -> 2048 ----
    hipMemsetAsync(agg, 0, (size_t)Mp * Fin * 4, stream);
    agg_edges_f32<<<E, 128, 0, stream>>>(x, src, dst, agg, E, Fin);
    mean_bf16_kernel<<<EW, 256, 0, stream>>>(agg, cnt, meanBf, N, Mp, Fin);
    sage_gemm<true><<<dim3(Mp / 32, H / 512), 256, 0, stream>>>(
        meanBf, xBf, W1lT, W1rT, b1, hA, Fin, H, Mp);

    // ---- layer 2: 2048 -> 2048 ----
    hipMemsetAsync(agg, 0, (size_t)Mp * H * 4, stream);
    agg_edges_bf16<<<E, 256, 0, stream>>>(hA, src, dst, agg, E, H);
    mean_bf16_kernel<<<EW, 256, 0, stream>>>(agg, cnt, meanBf, N, Mp, H);
    sage_gemm<true><<<dim3(Mp / 32, H / 512), 256, 0, stream>>>(
        meanBf, hA, W2lT, W2rT, b2, hB, H, H, Mp);

    // ---- layer 3: 2048 -> 1024, output f32 to d_out ----
    hipMemsetAsync(agg, 0, (size_t)Mp * H * 4, stream);
    agg_edges_bf16<<<E, 256, 0, stream>>>(hB, src, dst, agg, E, H);
    mean_bf16_kernel<<<EW, 256, 0, stream>>>(agg, cnt, meanBf, N, Mp, H);
    sage_gemm<false><<<dim3(Mp / 32, Fout / 512), 256, 0, stream>>>(
        meanBf, hB, W3lT, W3rT, b3, d_out, H, Fout, N);

    // ---- L2 normalize rows of d_out ----
    l2norm_kernel<<<N, 256, 0, stream>>>((float*)d_out, Fout);
}
